// RNN_87316685128405
// MI455X (gfx1250) — matrix operand order repaired and checked by reading, not compiled, not run on hardware
//
#include <hip/hip_runtime.h>
#include <hip/hip_bf16.h>
#include <math.h>

typedef __attribute__((ext_vector_type(16))) _Float16 v16h;
typedef __attribute__((ext_vector_type(8)))  float    v8f;
typedef long long i64;

#define BB 16
#define TT 20
#define TIN 10
#define HID 64
#define FCc 16
#define HH 64
#define WW 64
#define H4 16
#define W4 16
#define NN 256        // H4*W4
#define TFUT 10
#define NLAY 4

static __device__ __forceinline__ int imin(int a,int b){return a<b?a:b;}
static __device__ __forceinline__ int imax(int a,int b){return a>b?a:b;}

// ---------------------------------------------------------------------------
// Generic WMMA GEMM:  C(MxN) = act( beta*C + A * B^eff + bias )
//  a_mode 0: A[m*lda+k]   a_mode 1: A[k*lda+m]
//  b_mode 0: B[n*ldb+k]   b_mode 1: B[k*ldb+n]
//  act: 0 none, 1 relu, 2 sigmoid, 3 tanh.
//  4 waves/block along M; each wave register-blocks 4 N-tiles (16x64 strip):
//  one A fragment feeds 4 back-to-back v_wmma ops.  All addressing is
//  uniform-base + 32-bit element offsets (low VGPR pressure); contiguous-K
//  operands use float4 (b128) loads; masked tail chunk handles K%32.
// ---------------------------------------------------------------------------
__global__ __launch_bounds__(128)
void gemm_wmma(const float* __restrict__ A, int a_mode, int lda, i64 aS,
               const float* __restrict__ Bm, int b_mode, int ldb, i64 bS,
               const float* __restrict__ bias, int act, int beta,
               float* __restrict__ C, int ldc, i64 cS,
               int M, int N, int K)
{
  const int wave = threadIdx.x >> 5;
  const int lane = threadIdx.x & 31;
  const int half = lane >> 4;
  const int l16  = lane & 15;
  const int tm   = blockIdx.x * 4 + wave;
  const float* Ap = A  + (i64)blockIdx.z * aS;   // uniform (SGPR) bases
  const float* Bp = Bm + (i64)blockIdx.z * bS;
  float*       Cp = C  + (i64)blockIdx.z * cS;
  const int row  = tm * 16 + l16;
  const int rowc = imin(row, M - 1);
  const float rowMask = (row < M) ? 1.f : 0.f;
  const int aOff  = (a_mode == 0) ? rowc * lda : rowc;   // 32-bit offsets
  const int aStep = (a_mode == 0) ? 1 : lda;
  const int bStep = (b_mode == 0) ? 1 : ldb;
  const bool avec = (a_mode == 0) && ((lda & 3) == 0);
  const bool bvec = (b_mode == 0) && ((ldb & 3) == 0);
  const int kbA = half * 16;   // A f16 frag: k = kbA + e (contiguous 16)
  const int kbB = half * 16;   // B f16 frag: k = kbB + e (contiguous 16)

  int   col[4], bOff[4];
  float colMask[4];
#pragma unroll
  for (int j = 0; j < 4; ++j) {
    col[j] = blockIdx.y * 64 + j * 16 + l16;
    int cc = imin(col[j], N - 1);
    colMask[j] = (col[j] < N) ? 1.f : 0.f;
    bOff[j] = (b_mode == 0) ? cc * ldb : cc;
  }

  v8f acc[4] = {};
  if (beta) {
    const int mb = tm * 16 + half * 8;
#pragma unroll
    for (int j = 0; j < 4; ++j) {
      int cc = imin(col[j], N - 1);
#pragma unroll
      for (int r = 0; r < 8; ++r) {
        int m = mb + r;
        float msk = (m < M) ? colMask[j] : 0.f;
        acc[j][r] = Cp[imin(m, M - 1) * ldc + cc] * msk;
      }
    }
  }

  const int Km = K & ~31;
  for (int kc = 0; kc < Km; kc += 32) {
    v16h a;
    if (avec) {
      const float4* pa = (const float4*)(Ap + (aOff + kc + kbA));
      float va[16];
      *(float4*)&va[0]  = pa[0];
      *(float4*)&va[4]  = pa[1];
      *(float4*)&va[8]  = pa[2];   // +8 floats
      *(float4*)&va[12] = pa[3];
#pragma unroll
      for (int e = 0; e < 16; ++e) a[e] = (_Float16)(va[e] * rowMask);
    } else {
#pragma unroll
      for (int e = 0; e < 16; ++e) {
        int ka = kc + kbA + e;
        a[e] = (_Float16)(Ap[aOff + ka * aStep] * rowMask);
      }
    }
#pragma unroll
    for (int j = 0; j < 4; ++j) {
      v16h b;
      if (bvec) {
        const float4* pb = (const float4*)(Bp + (bOff[j] + kc + kbB));
        float vb[16];
        *(float4*)&vb[0]  = pb[0];
        *(float4*)&vb[4]  = pb[1];
        *(float4*)&vb[8]  = pb[2];
        *(float4*)&vb[12] = pb[3];
#pragma unroll
        for (int e = 0; e < 16; ++e) b[e] = (_Float16)(vb[e] * colMask[j]);
      } else {
#pragma unroll
        for (int e = 0; e < 16; ++e) {
          int kb = kc + kbB + e;
          b[e] = (_Float16)(Bp[bOff[j] + kb * bStep] * colMask[j]);
        }
      }
      acc[j] = __builtin_amdgcn_wmma_f32_16x16x32_f16(false, a, false, b,
                                                      (short)0, acc[j], false, false);
    }
  }
  if (Km < K) {            // masked tail chunk (K % 32 != 0)
    v16h a;
#pragma unroll
    for (int e = 0; e < 16; ++e) {
      int ka = Km + kbA + e;
      float am = (ka < K) ? rowMask : 0.f;
      a[e] = (_Float16)(Ap[aOff + imin(ka, K - 1) * aStep] * am);
    }
#pragma unroll
    for (int j = 0; j < 4; ++j) {
      v16h b;
#pragma unroll
      for (int e = 0; e < 16; ++e) {
        int kb = Km + kbB + e;
        float bm = (kb < K) ? colMask[j] : 0.f;
        b[e] = (_Float16)(Bp[bOff[j] + imin(kb, K - 1) * bStep] * bm);
      }
      acc[j] = __builtin_amdgcn_wmma_f32_16x16x32_f16(false, a, false, b,
                                                      (short)0, acc[j], false, false);
    }
  }

  const int mb = tm * 16 + half * 8;
#pragma unroll
  for (int j = 0; j < 4; ++j) {
    float bv = (bias != nullptr && col[j] < N) ? bias[col[j]] : 0.f;
#pragma unroll
    for (int r = 0; r < 8; ++r) {
      int m = mb + r;
      if (m < M && col[j] < N) {
        float v = acc[j][r] + bv;
        if      (act == 1) v = fmaxf(v, 0.f);
        else if (act == 2) v = 1.f / (1.f + __expf(-v));
        else if (act == 3) v = tanhf(v);
        Cp[m * ldc + col[j]] = v;
      }
    }
  }
}

// ---------------------------------------------------------------------------
// Implicit-GEMM 3x3 conv via WMMA.  M = outH*outW, N = Cout, K = Cin*9.
//  nhwc: stem-frames input layout.  up2f: fused 2x nearest upsample gather.
//  csplit: channels >= csplit read from in2 (fused concat) -- selected
//  *uniformly per K-chunk* (split k = csplit*9 is 32-aligned for all users).
//  res: fused residual add.  4-wide N blocking; float4 weight loads;
//  uniform-base + 32-bit offset addressing.
// ---------------------------------------------------------------------------
__global__ __launch_bounds__(128)
void conv3x3_wmma(const float* __restrict__ in,  i64 inS,
                  const float* __restrict__ in2, i64 in2S, int csplit,
                  int nhwc, int inH, int inW, int Cin,
                  int outH, int outW, int stride, int pad, int up2f,
                  const float* __restrict__ w,
                  const float* __restrict__ bias,
                  const float* __restrict__ res, i64 resS,
                  int act,
                  float* __restrict__ out, i64 outS, int Cout)
{
  const int M = outH * outW;
  const int K = Cin * 9;
  const int kSplit = csplit * 9;
  const int img  = blockIdx.z;
  const int wave = threadIdx.x >> 5;
  const int lane = threadIdx.x & 31;
  const int half = lane >> 4;
  const int l16  = lane & 15;
  const int tm   = blockIdx.x * 4 + wave;
  const int row  = tm * 16 + l16;
  const int rowc = imin(row, M - 1);
  const float rowMask = (row < M) ? 1.f : 0.f;
  const int oy = rowc / outW, ox = rowc % outW;
  const int eH = up2f ? inH * 2 : inH;
  const int eW = up2f ? inW * 2 : inW;
  const float* pin  = in  + (i64)img * inS;    // uniform bases
  const float* pin2 = in2 + (i64)img * in2S;
  const int kbA = half * 16;
  const int kbB = half * 16;

  int   col[4], wOff[4];
  float colMask[4];
#pragma unroll
  for (int j = 0; j < 4; ++j) {
    col[j] = blockIdx.y * 64 + j * 16 + l16;
    colMask[j] = (col[j] < Cout) ? 1.f : 0.f;
    wOff[j] = imin(col[j], Cout - 1) * K;      // OIHW row offset
  }

  // 32-bit offset + validity mask for one (cin,kh,kw) gather
  auto fetchA = [&](int ka, const float* pu, int cAdj) -> float {
    int cin = ka / 9, kk = ka - cin * 9;
    int kh = kk / 3, kw = kk - kh * 3;
    int iy = oy * stride + kh - pad;
    int ix = ox * stride + kw - pad;
    float ok = (iy >= 0 && iy < eH && ix >= 0 && ix < eW) ? rowMask : 0.f;
    int sy = up2f ? (iy >> 1) : iy;
    int sx = up2f ? (ix >> 1) : ix;
    sy = imin(imax(sy, 0), inH - 1);
    sx = imin(imax(sx, 0), inW - 1);
    int c = cin - cAdj;
    int off = nhwc ? (sy * inW + sx) * Cin + c
                   : c * inH * inW + sy * inW + sx;
    return pu[off] * ok;
  };

  v8f acc[4] = {};
  const int Km = K & ~31;
  for (int kc = 0; kc < Km; kc += 32) {
    const float* pu = (kc >= kSplit) ? pin2 : pin;   // scalar per chunk
    const int  cAdj = (kc >= kSplit) ? csplit : 0;
    v16h a;
#pragma unroll
    for (int e = 0; e < 16; ++e) {
      int ka = kc + kbA + e;
      a[e] = (_Float16)fetchA(ka, pu, cAdj);
    }
#pragma unroll
    for (int j = 0; j < 4; ++j) {
      // weights contiguous in K (Cin*9 is a multiple of 4 for all layers)
      const float4* pw = (const float4*)(w + (wOff[j] + kc + kbB));
      float vb[16];
      *(float4*)&vb[0]  = pw[0];
      *(float4*)&vb[4]  = pw[1];
      *(float4*)&vb[8]  = pw[2];
      *(float4*)&vb[12] = pw[3];
      v16h b;
#pragma unroll
      for (int e = 0; e < 16; ++e) b[e] = (_Float16)(vb[e] * colMask[j]);
      acc[j] = __builtin_amdgcn_wmma_f32_16x16x32_f16(false, a, false, b,
                                                      (short)0, acc[j], false, false);
    }
  }
  if (Km < K) {            // masked tail (stem conv1: K=144)
    const float* pu = (Km >= kSplit) ? pin2 : pin;
    const int  cAdj = (Km >= kSplit) ? csplit : 0;
    v16h a;
#pragma unroll
    for (int e = 0; e < 16; ++e) {
      int ka = Km + kbA + e;
      a[e] = (_Float16)((ka < K) ? fetchA(ka, pu, cAdj) : 0.f);
    }
#pragma unroll
    for (int j = 0; j < 4; ++j) {
      v16h b;
#pragma unroll
      for (int e = 0; e < 16; ++e) {
        int kb = Km + kbB + e;
        float bm = (kb < K) ? colMask[j] : 0.f;
        b[e] = (_Float16)(w[wOff[j] + imin(kb, K - 1)] * bm);
      }
      acc[j] = __builtin_amdgcn_wmma_f32_16x16x32_f16(false, a, false, b,
                                                      (short)0, acc[j], false, false);
    }
  }

  const int mb = tm * 16 + half * 8;
  float* op = out + (i64)img * outS;
  const float* rp = res ? res + (i64)img * resS : nullptr;
#pragma unroll
  for (int j = 0; j < 4; ++j) {
    float bs = (bias != nullptr && col[j] < Cout) ? bias[col[j]] : 0.f;
#pragma unroll
    for (int r = 0; r < 8; ++r) {
      int m = mb + r;
      if (m < M && col[j] < Cout) {
        float v = acc[j][r] + bs;
        if (rp) v += rp[col[j] * M + m];
        if (act == 1) v = fmaxf(v, 0.f);
        op[col[j] * M + m] = v;               // NCHW
      }
    }
  }
}

// ---------------------------------------------------------------------------
// Row softmax, one wave32 per row.
// ---------------------------------------------------------------------------
__global__ __launch_bounds__(32)
void softmax_rows(float* __restrict__ d, int width)
{
  float* row = d + (i64)blockIdx.x * width;
  int lane = threadIdx.x;
  float mx = -3.0e38f;
  for (int j = lane; j < width; j += 32) mx = fmaxf(mx, row[j]);
  for (int o = 16; o > 0; o >>= 1) mx = fmaxf(mx, __shfl_xor(mx, o));
  float s = 0.f;
  for (int j = lane; j < width; j += 32) {
    float e = __expf(row[j] - mx); row[j] = e; s += e;
  }
  for (int o = 16; o > 0; o >>= 1) s += __shfl_xor(s, o);
  float inv = 1.f / s;
  for (int j = lane; j < width; j += 32) row[j] *= inv;
}

// dst (C x R) = transpose of src (R x C)   -- used once for Wzr/Wc/M
__global__ void transpose_kernel(const float* __restrict__ src,
                                 float* __restrict__ dst, int R, int Ccols)
{
  i64 idx = (i64)blockIdx.x * blockDim.x + threadIdx.x;
  if (idx >= (i64)R * Ccols) return;
  int r = (int)(idx / Ccols), c = (int)(idx % Ccols);
  dst[(i64)c * R + r] = src[idx];
}

// net_seq[:,t] = t<TIN-2 ? fd[:,t] : m*fd[:,t] + (1-m)*ode[(t-9)%10]
__global__ void netseq_kernel(const float* __restrict__ fd,
                              const float* __restrict__ ode,
                              const float* __restrict__ mask,
                              float* __restrict__ net)
{
  i64 idx = (i64)blockIdx.x * blockDim.x + threadIdx.x;
  if (idx >= (i64)BB * 19 * HID * NN) return;
  int b = (int)(idx / (19 * HID * NN));
  int r = (int)(idx % (19 * HID * NN));
  int t = r / (HID * NN);
  int e = r % (HID * NN);
  float fdv = fd[((i64)(b * TT + t)) * HID * NN + e];
  float v;
  if (t < TIN - 2) v = fdv;
  else {
    int mi = t - 1;
    float m = mask[(i64)(b * 19 + mi) * (HH * WW)];
    int os = ((t - (TIN - 1)) % TFUT + TFUT) % TFUT;
    float ov = ode[((i64)(b * TFUT + os)) * HID * NN + e];
    v = m * fdv + (1.f - m) * ov;
  }
  net[idx] = v;
}

// inpT[b][c][m]  (channel-major!)  c<64: xf, c<128: val, else (r*)hf
__global__ void build_xin(const float* __restrict__ X, i64 xS,
                          const float* __restrict__ val,
                          const float* __restrict__ Hs,
                          const float* __restrict__ zr,
                          float* __restrict__ inpT)
{
  i64 idx = (i64)blockIdx.x * blockDim.x + threadIdx.x;
  if (idx >= (i64)BB * 192 * NN) return;
  int b = (int)(idx / (192 * NN));
  int r = (int)(idx % (192 * NN));
  int c = r / NN, m = r % NN;
  float v;
  if (c < HID)          v = X[(i64)b * xS + c * NN + m];
  else if (c < 2 * HID) v = val[(i64)b * NN * HID + m * HID + (c - HID)];
  else {
    int cc = c - 2 * HID;
    float h = Hs[(i64)b * HID * NN + cc * NN + m];
    if (zr) h *= zr[(i64)b * NN * 128 + m * 128 + 64 + cc];
    v = h;
  }
  inpT[idx] = v;
}

// H = z*H + (1-z)*hc
__global__ void update_h(float* __restrict__ Hs,
                         const float* __restrict__ zr,
                         const float* __restrict__ hc)
{
  i64 idx = (i64)blockIdx.x * blockDim.x + threadIdx.x;
  if (idx >= (i64)BB * HID * NN) return;
  int b = (int)(idx / (HID * NN));
  int r = (int)(idx % (HID * NN));
  int c = r / NN, m = r % NN;
  float z  = zr[(i64)b * NN * 128 + m * 128 + c];
  float hv = hc[(i64)b * NN * HID + m * HID + c];
  Hs[idx] = z * Hs[idx] + (1.f - z) * hv;
}

__global__ void init_last(const float* __restrict__ frames, float* __restrict__ last)
{
  i64 idx = (i64)blockIdx.x * blockDim.x + threadIdx.x;
  if (idx >= (i64)BB * FCc * HH * WW) return;
  int b = (int)(idx / (FCc * HH * WW));
  int r = (int)(idx % (FCc * HH * WW));
  int c = r / (HH * WW), p = r % (HH * WW);
  last[idx] = frames[((i64)b * TT * HH * WW + p) * FCc + c];
}

__device__ __forceinline__ void wave_atomic(float v, float* dst)
{
  for (int o = 16; o > 0; o >>= 1) v += __shfl_xor(v, o);
  if ((threadIdx.x & 31) == 0) atomicAdd(dst, v);
}

// Bilinear border warp + blend + next_frames + fused mse/inter losses
__global__ void warp_pred_kernel(const float* __restrict__ frames,
                                 const float* __restrict__ d3,
                                 const float* __restrict__ last,
                                 float* __restrict__ cur,
                                 float* __restrict__ outp,
                                 float* __restrict__ accum, int t)
{
  i64 idx = (i64)blockIdx.x * blockDim.x + threadIdx.x;
  float mse = 0.f, il = 0.f;
  if (idx < (i64)BB * HH * WW) {
    int b = (int)(idx >> 12);
    int p = (int)(idx & 4095);
    int y = p >> 6, x = p & 63;
    const float* db = d3 + (i64)b * 19 * (HH * WW);
    float fx = db[p], fy = db[HH * WW + p];
    float gx = (-1.f + 2.f * x / 63.f) + fx / 31.5f;
    float gy = (-1.f + 2.f * y / 63.f) + fy / 31.5f;
    float xg = (gx + 1.f) * 32.f - 0.5f;
    float yg = (gy + 1.f) * 32.f - 0.5f;
    float x0f = floorf(xg), y0f = floorf(yg);
    float wx = xg - x0f, wy = yg - y0f;
    int x0 = imin(imax((int)x0f, 0), 63), x1 = imin(imax((int)x0f + 1, 0), 63);
    int y0 = imin(imax((int)y0f, 0), 63), y1 = imin(imax((int)y0f + 1, 0), 63);
    float mk = 1.f / (1.f + __expf(-db[18 * HH * WW + p]));
    const float* lb = last + (i64)b * FCc * HH * WW;
    for (int c = 0; c < FCc; ++c) {
      const float* ch = lb + c * HH * WW;
      float v00 = ch[y0 * 64 + x0], v01 = ch[y0 * 64 + x1];
      float v10 = ch[y1 * 64 + x0], v11 = ch[y1 * 64 + x1];
      float wv = v00 * (1.f - wx) * (1.f - wy) + v01 * wx * (1.f - wy)
               + v10 * (1.f - wx) * wy          + v11 * wx * wy;
      float in_c = db[(2 + c) * HH * WW + p];
      float pr = mk * wv + (1.f - mk) * in_c;
      cur[(i64)b * FCc * HH * WW + c * HH * WW + p] = wv;
      outp[(((i64)(b * 19 + t)) * (HH * WW) + p) * FCc + c] = pr;
      float fn = frames[(((i64)(b * TT + t + 1)) * (HH * WW) + p) * FCc + c];
      float fc = frames[(((i64)(b * TT + t)) * (HH * WW) + p) * FCc + c];
      mse += (pr - fn) * (pr - fn);
      float dr = fn - fc;
      il += (dr - in_c) * (dr - in_c);
    }
  }
  wave_atomic(mse, &accum[0]);
  wave_atomic(il,  &accum[1]);
}

__global__ void ode_loss_kernel(const float* __restrict__ ode,
                                const float* __restrict__ fd,
                                float* __restrict__ accum)
{
  i64 idx = (i64)blockIdx.x * blockDim.x + threadIdx.x;
  float v = 0.f;
  if (idx < (i64)BB * TFUT * HID * NN) {
    int b = (int)(idx / (TFUT * HID * NN));
    int r = (int)(idx % (TFUT * HID * NN));
    int s = r / (HID * NN), e = r % (HID * NN);
    float d = ode[idx] - fd[((i64)(b * TT + TIN + s)) * HID * NN + e];
    v = d * d;
  }
  wave_atomic(v, &accum[2]);
}

__global__ void finalize_kernel(const float* __restrict__ accum,
                                float* __restrict__ outp, int out_size)
{
  if (blockIdx.x == 0 && threadIdx.x == 0) {
    const float invm = 1.f / ((float)BB * 19 * HH * WW * FCc);
    const float invo = 1.f / ((float)BB * TFUT * HID * NN);
    outp[out_size - 1] = accum[0] * invm + accum[1] * invm + 0.03f * accum[2] * invo;
  }
}

// ===========================================================================
extern "C" void kernel_launch(void* const* d_in, const int* in_sizes, int n_in,
                              void* d_out, int out_size, void* d_ws, size_t ws_size,
                              hipStream_t stream)
{
  const float* frames = (const float*)d_in[0];   // (B,T,H,W,FC)
  const float* maskp  = (const float*)d_in[1];   // (B,19,H,W,1)
  const float* stem_w1 = (const float*)d_in[2];
  const float* stem_b1 = (const float*)d_in[3];
  const float* stem_w2 = (const float*)d_in[4];
  const float* stem_b2 = (const float*)d_in[5];
  const float* dec_w1  = (const float*)d_in[6];
  const float* dec_b1  = (const float*)d_in[7];
  const float* dec_w2  = (const float*)d_in[8];
  const float* dec_b2  = (const float*)d_in[9];
  const float* dec_w3  = (const float*)d_in[10];
  const float* dec_b3  = (const float*)d_in[11];
  const float* ode_w1  = (const float*)d_in[12];
  const float* ode_b1  = (const float*)d_in[13];
  const float* ode_w2  = (const float*)d_in[14];
  const float* ode_b2  = (const float*)d_in[15];
  const float* last_w  = (const float*)d_in[16];  // (64,64,1,1)
  const float* cell_M  = (const float*)d_in[17];  // (4,32,64)
  const float* cell_E  = (const float*)d_in[18];  // (4,256,10)
  const float* cell_Wzr= (const float*)d_in[19];  // (4,384,128)
  const float* cell_bzr= (const float*)d_in[20];  // (4,128)
  const float* cell_Wc = (const float*)d_in[21];  // (4,384,64)
  const float* cell_bc = (const float*)d_in[22];  // (4,64)
  float* outp = (float*)d_out;

  // ---- workspace carve (f32 elements) ----
  float* Wb = (float*)d_ws;
  size_t off = 0;
  auto alloc = [&](size_t n) { float* p = Wb + off; off += n; return p; };
  const i64 IM = (i64)HID * NN;                 // 16384
  float* fd    = alloc((size_t)BB * TT * IM);
  float* s1    = alloc((size_t)BB * HID * 32 * 32);
  float* ode   = alloc((size_t)BB * TFUT * IM);
  float* odeH  = alloc((size_t)BB * 128 * NN);
  float* net   = alloc((size_t)BB * 19 * IM);
  float* Amat  = alloc((size_t)NLAY * NN * NN);
  float* hstate= alloc((size_t)NLAY * BB * IM);
  float* att   = alloc((size_t)BB * NN * 32);
  float* val   = alloc((size_t)BB * NN * HID);
  float* inpT  = alloc((size_t)BB * 192 * NN);
  float* ginp  = alloc((size_t)BB * NN * 192);
  float* zr    = alloc((size_t)BB * NN * 128);
  float* hc    = alloc((size_t)BB * NN * HID);
  float* sol   = alloc((size_t)BB * 19 * IM);
  float* d1    = alloc((size_t)BB * 128 * NN);
  float* d2    = alloc((size_t)BB * HID * 32 * 32);
  float* d3    = alloc((size_t)BB * 19 * HH * WW);
  float* lastA = alloc((size_t)BB * FCc * HH * WW);
  float* lastB = alloc((size_t)BB * FCc * HH * WW);
  float* WzrT  = alloc((size_t)NLAY * 128 * 384);   // (128 x 384) per layer
  float* WcT   = alloc((size_t)NLAY * 64 * 384);    // (64 x 384)  per layer
  float* MT    = alloc((size_t)NLAY * 64 * 32);     // (64 x 32)   per layer
  float* accum = alloc(8);

  hipMemsetAsync(hstate, 0, (size_t)NLAY * BB * IM * sizeof(float), stream);
  hipMemsetAsync(accum, 0, 8 * sizeof(float), stream);

  auto gemm = [&](const float* A, int am, int lda, i64 aS,
                  const float* Bm, int bm, int ldb, i64 bS,
                  const float* bias, int act, int beta,
                  float* C, int ldc, i64 cS, int M, int N, int K, int batch) {
    dim3 g((M + 63) / 64, (N + 63) / 64, batch);
    gemm_wmma<<<g, dim3(128), 0, stream>>>(A, am, lda, aS, Bm, bm, ldb, bS,
                                           bias, act, beta, C, ldc, cS, M, N, K);
  };
  auto conv = [&](const float* in, i64 inS, const float* in2, i64 in2S, int csplit,
                  int nhwc, int inH, int inW, int Cin, int oH, int oW,
                  int stride, int pad, int up2f, const float* w, const float* bias,
                  const float* res, i64 resS, int act,
                  float* out, i64 outS, int Cout, int batch) {
    dim3 g((oH * oW + 63) / 64, (Cout + 63) / 64, batch);
    conv3x3_wmma<<<g, dim3(128), 0, stream>>>(in, inS, in2, in2S, csplit, nhwc,
        inH, inW, Cin, oH, oW, stride, pad, up2f, w, bias, res, resS, act,
        out, outS, Cout);
  };
  auto ew = [&](i64 n) { return dim3((unsigned)((n + 255) / 256)); };

  // ---- one-time weight transposes (K-major -> N-major for vector loads) ----
  for (int i = 0; i < NLAY; ++i) {
    transpose_kernel<<<ew(384 * 128), 256, 0, stream>>>(
        cell_Wzr + (i64)i * 384 * 128, WzrT + (i64)i * 128 * 384, 384, 128);
    transpose_kernel<<<ew(384 * 64), 256, 0, stream>>>(
        cell_Wc + (i64)i * 384 * 64, WcT + (i64)i * 64 * 384, 384, 64);
    transpose_kernel<<<ew(32 * 64), 256, 0, stream>>>(
        cell_M + (i64)i * 32 * 64, MT + (i64)i * 64 * 32, 32, 64);
  }

  // ---- stem: 320 frames, chunked per t (B images) ----
  for (int t = 0; t < TT; ++t) {
    conv(frames + (i64)t * HH * WW * FCc, (i64)TT * HH * WW * FCc,
         frames, (i64)TT * HH * WW * FCc, FCc,
         1, HH, WW, FCc, 32, 32, 2, 0, 0,
         stem_w1, stem_b1, nullptr, 0, 1, s1, (i64)HID * 1024, HID, BB);
    conv(s1, (i64)HID * 1024, s1, (i64)HID * 1024, HID,
         0, 32, 32, HID, H4, W4, 2, 0, 0,
         stem_w2, stem_b2, nullptr, 0, 1,
         fd + (i64)t * IM, (i64)TT * IM, HID, BB);
  }

  // ---- ODE block: x_{s+1} = x_s + conv2(relu(conv1(x_s))) ----
  for (int s = 0; s < TFUT; ++s) {
    const float* xp = s ? ode + (i64)(s - 1) * IM : fd + (i64)(TIN - 1) * IM;
    i64 xs = s ? (i64)TFUT * IM : (i64)TT * IM;
    conv(xp, xs, xp, xs, HID, 0, H4, W4, HID, H4, W4, 1, 1, 0,
         ode_w1, ode_b1, nullptr, 0, 1, odeH, (i64)128 * NN, 128, BB);
    conv(odeH, (i64)128 * NN, odeH, (i64)128 * NN, 128,
         0, H4, W4, 128, H4, W4, 1, 1, 0,
         ode_w2, ode_b2, xp, xs, 0, ode + (i64)s * IM, (i64)TFUT * IM, HID, BB);
  }
  ode_loss_kernel<<<ew((i64)BB * TFUT * IM), 256, 0, stream>>>(ode, fd, accum);

  // ---- net_seq blend ----
  netseq_kernel<<<ew((i64)BB * 19 * IM), 256, 0, stream>>>(fd, ode, maskp, net);

  // ---- per-layer graph adjacency A = softmax(relu(E E^T)) ----
  for (int i = 0; i < NLAY; ++i) {
    const float* Ei = cell_E + (i64)i * NN * 10;
    gemm(Ei, 0, 10, 0, Ei, 0, 10, 0, nullptr, 1, 0,
         Amat + (i64)i * NN * NN, NN, 0, NN, NN, 10, 1);
    softmax_rows<<<NN, 32, 0, stream>>>(Amat + (i64)i * NN * NN, NN);
  }

  // ---- recurrent loop ----
  for (int t = 0; t < TT - 1; ++t) {
    for (int i = 0; i < NLAY; ++i) {
      const float* Xp = (i == 0) ? net + (i64)t * IM : hstate + (i64)(i - 1) * BB * IM;
      i64 xS = (i == 0) ? (i64)19 * IM : IM;
      float* Hp = hstate + (i64)i * BB * IM;
      const float* Mi   = cell_M + (i64)i * 32 * HID;
      const float* MTi  = MT   + (i64)i * 64 * 32;
      const float* WzrTi= WzrT + (i64)i * 128 * 384;
      const float* WcTi = WcT  + (i64)i * 64 * 384;
      const float* Ai   = Amat + (i64)i * NN * NN;

      // attention read of memory matrix
      gemm(Hp, 1, NN, IM, Mi, 0, HID, 0, nullptr, 0, 0,
           att, 32, (i64)NN * 32, NN, 32, HID, BB);
      softmax_rows<<<BB * NN, 32, 0, stream>>>(att, 32);
      gemm(att, 0, 32, (i64)NN * 32, MTi, 0, 32, 0, nullptr, 0, 0,
           val, HID, (i64)NN * HID, NN, HID, 32, BB);

      // gates: sup @ Wzr = inp@W[:192] + (A@inp)@W[192:]
      build_xin<<<ew((i64)BB * NN * 192), 256, 0, stream>>>(Xp, xS, val, Hp, nullptr, inpT);
      gemm(Ai, 0, NN, 0, inpT, 0, NN, (i64)192 * NN, nullptr, 0, 0,
           ginp, 192, (i64)NN * 192, NN, 192, NN, BB);
      gemm(inpT, 1, NN, (i64)192 * NN, WzrTi, 0, 384, 0,
           cell_bzr + (i64)i * 128, 0, 0, zr, 128, (i64)NN * 128, NN, 128, 192, BB);
      gemm(ginp, 0, 192, (i64)NN * 192, WzrTi + 192, 0, 384, 0,
           nullptr, 2, 1, zr, 128, (i64)NN * 128, NN, 128, 192, BB);

      // candidate with reset gate
      build_xin<<<ew((i64)BB * NN * 192), 256, 0, stream>>>(Xp, xS, val, Hp, zr, inpT);
      gemm(Ai, 0, NN, 0, inpT, 0, NN, (i64)192 * NN, nullptr, 0, 0,
           ginp, 192, (i64)NN * 192, NN, 192, NN, BB);
      gemm(inpT, 1, NN, (i64)192 * NN, WcTi, 0, 384, 0,
           cell_bc + (i64)i * HID, 0, 0, hc, HID, (i64)NN * HID, NN, HID, 192, BB);
      gemm(ginp, 0, 192, (i64)NN * 192, WcTi + 192, 0, 384, 0,
           nullptr, 3, 1, hc, HID, (i64)NN * HID, NN, HID, 192, BB);

      update_h<<<ew((i64)BB * IM), 256, 0, stream>>>(Hp, zr, hc);
    }
    // gen = 1x1 conv(top state) -> sol_y[:,t] (C rows = Cout -> NCHW)
    gemm(last_w, 0, HID, 0, hstate + (i64)3 * BB * IM, 1, NN, IM,
         nullptr, 0, 0, sol + (i64)t * IM, NN, (i64)19 * IM, HID, NN, HID, BB);
  }

  // ---- decoder + warp scan ----
  init_last<<<ew((i64)BB * FCc * HH * WW), 256, 0, stream>>>(frames, lastA);
  float* lastP = lastA;
  float* curP  = lastB;
  for (int t = 0; t < TT - 1; ++t) {
    const float* prev = t ? sol + (i64)(t - 1) * IM : fd;
    i64 prevS = t ? (i64)19 * IM : (i64)TT * IM;
    conv(sol + (i64)t * IM, (i64)19 * IM, prev, prevS, HID,
         0, H4, W4, 2 * HID, H4, W4, 1, 1, 0,
         dec_w1, dec_b1, nullptr, 0, 1, d1, (i64)128 * NN, 128, BB);
    conv(d1, (i64)128 * NN, d1, (i64)128 * NN, 128,
         0, H4, W4, 128, 32, 32, 1, 1, 1,
         dec_w2, dec_b2, nullptr, 0, 1, d2, (i64)HID * 1024, HID, BB);
    conv(d2, (i64)HID * 1024, d2, (i64)HID * 1024, HID,
         0, 32, 32, HID, HH, WW, 1, 1, 1,
         dec_w3, dec_b3, nullptr, 0, 0, d3, (i64)19 * HH * WW, FCc + 3, BB);
    warp_pred_kernel<<<ew((i64)BB * HH * WW), 256, 0, stream>>>(
        frames, d3, lastP, curP, outp, accum, t);
    float* tmp = lastP; lastP = curP; curP = tmp;
  }

  finalize_kernel<<<1, 32, 0, stream>>>(accum, outp, out_size);
}
